// _MolecularGNNEncoder_85409719648816
// MI455X (gfx1250) — matrix operand order repaired and checked by reading, not compiled, not run on hardware
//
#include <hip/hip_runtime.h>
#include <hip/hip_bf16.h>

typedef _Float16 f16;
typedef __attribute__((ext_vector_type(16))) _Float16 v16h;
typedef __attribute__((ext_vector_type(8)))  _Float16 v8h;
typedef __attribute__((ext_vector_type(8)))  float    v8f;

#define N_NODES  100000
#define N_EDGES  1600000
#define N_GRAPHS 4096
#define K_IN     128
#define BN_EPS   1e-5f
#define TPB      256

static __device__ __forceinline__ int gstride() { return gridDim.x * blockDim.x; }

// ---------- small utility kernels ----------
__global__ void k_fill_f32(float* __restrict__ p, float v, int n) {
    for (int i = blockIdx.x * blockDim.x + threadIdx.x; i < n; i += gstride()) p[i] = v;
}

__global__ void k_deg(const int* __restrict__ tgt, float* __restrict__ deg) {
    for (int e = blockIdx.x * blockDim.x + threadIdx.x; e < N_EDGES; e += gstride())
        atomicAdd(&deg[tgt[e]], 1.0f);
}

__global__ void k_rsqrt_inplace(float* __restrict__ d, int n) {
    for (int i = blockIdx.x * blockDim.x + threadIdx.x; i < n; i += gstride())
        d[i] = rsqrtf(d[i]);
}

__global__ void k_cvt_f16(const float* __restrict__ in, f16* __restrict__ out, int n) {
    for (int i = blockIdx.x * blockDim.x + threadIdx.x; i < n; i += gstride())
        out[i] = (f16)in[i];
}

// W: [K_IN, Fout] row-major f32  ->  Wt: [Fout, K_IN] row-major f16
__global__ void k_wt_cvt(const float* __restrict__ W, f16* __restrict__ Wt, int Fout) {
    int total = K_IN * Fout;
    for (int idx = blockIdx.x * blockDim.x + threadIdx.x; idx < total; idx += gstride()) {
        int k = idx / Fout, n = idx % Fout;
        Wt[n * K_IN + k] = (f16)W[idx];
    }
}

// ---------- WMMA GEMM with LDS-staged weights + fused self-loop/bias epilogue ----------
// hw[M,FOUT]  = A[M,K_IN](f16) x Wt[FOUT,K_IN]^T(f16)   (f32 accumulate)
// agg[M,FOUT] = hw * dinv[i]^2 + bias[f]                (GCN self-loop + bias init)
// One wave owns a 16-row strip and sweeps all FOUT/16 N-tiles, reusing the four
// K-step A fragments held in VGPRs. B fragments are served from LDS (ds_load_b128).
template <int FOUT>
__global__ void k_gemm_wmma_lds(const f16* __restrict__ A, const f16* __restrict__ Wt,
                                const float* __restrict__ dinv, const float* __restrict__ bias,
                                float* __restrict__ hw, float* __restrict__ agg, int M) {
    extern __shared__ f16 ldsW[];                       // FOUT x K_IN halves
    constexpr int TOT = FOUT * K_IN;
    for (int i = threadIdx.x * 8; i < TOT; i += blockDim.x * 8)
        *(v8h*)(ldsW + i) = *(const v8h*)(Wt + i);
    __syncthreads();

    int wave = threadIdx.x >> 5;
    int rowTile = blockIdx.x * 8 + wave;                // 16-row strip per wave
    if (rowTile * 16 >= M) return;                      // wave-uniform: EXEC stays all-1s

    int lane = threadIdx.x & 31;
    int r = lane & 15;                                  // row/col within tile
    int g = lane >> 4;                                  // lane half

    // Preload all four K-step A fragments (A 16x32 f16 layout:
    // lane half g holds K = kb + {8g..8g+7} and {16+8g..16+8g+7})
    const f16* arow = A + (size_t)(rowTile * 16 + r) * K_IN;
    v16h a[4];
    #pragma unroll
    for (int kb = 0; kb < 4; ++kb) {
        v8h alo = *(const v8h*)(arow + kb * 32 + 8 * g);
        v8h ahi = *(const v8h*)(arow + kb * 32 + 16 + 8 * g);
        #pragma unroll
        for (int j = 0; j < 8; ++j) { a[kb][j] = alo[j]; a[kb][j + 8] = ahi[j]; }
    }

    // Per-lane self-loop scales: D-matrix VGPR v <-> row M = rowTile*16 + v + 8g
    float d2[8];
    #pragma unroll
    for (int v = 0; v < 8; ++v) {
        float d = dinv[rowTile * 16 + 8 * g + v];
        d2[v] = d * d;
    }

    #pragma unroll
    for (int tn = 0; tn < FOUT / 16; ++tn) {
        const f16* bcol = ldsW + (size_t)(tn * 16 + r) * K_IN;   // col N = tn*16+r, contiguous K
        v8f c = {0.f, 0.f, 0.f, 0.f, 0.f, 0.f, 0.f, 0.f};
        #pragma unroll
        for (int kb = 0; kb < 4; ++kb) {
            // B 32x16 f16 layout: lane half g holds K = kb*32 + {8g..8g+7} and {16+8g..16+8g+7} for fixed N
            v8h blo = *(const v8h*)(bcol + kb * 32 + 8 * g);
            v8h bhi = *(const v8h*)(bcol + kb * 32 + 16 + 8 * g);
            v16h b;
            #pragma unroll
            for (int j = 0; j < 8; ++j) { b[j] = blo[j]; b[j + 8] = bhi[j]; }
            c = __builtin_amdgcn_wmma_f32_16x16x32_f16(false, a[kb], false, b,
                                                       (short)0, c, false, false);
        }
        float bval = bias[tn * 16 + r];
        size_t base = (size_t)(rowTile * 16 + 8 * g) * FOUT + tn * 16 + r;
        #pragma unroll
        for (int v = 0; v < 8; ++v) {
            float val = c[v];
            hw[base + (size_t)v * FOUT]  = val;
            agg[base + (size_t)v * FOUT] = val * d2[v] + bval;
        }
    }
}

// ---------- GCN edge aggregation: agg[tgt,f] += hw[src,f] * dinv[src]*dinv[tgt] ----------
__global__ void k_edge_scatter(const int* __restrict__ src, const int* __restrict__ tgt,
                               const float* __restrict__ hw, const float* __restrict__ dinv,
                               float* __restrict__ agg, int F) {
    int total = N_EDGES * F;   // <= 409.6M, fits int32
    for (int idx = blockIdx.x * blockDim.x + threadIdx.x; idx < total; idx += gstride()) {
        int e = idx / F, f = idx - e * F;
        int s = src[e], t = tgt[e];
        float nrm = dinv[s] * dinv[t];
        atomicAdd(&agg[t * F + f], hw[s * F + f] * nrm);
    }
}

// ---------- BatchNorm (F == 128) ----------
__global__ void k_bn_stats128(const float* __restrict__ x, float* __restrict__ stats) {
    __shared__ float s[128];
    __shared__ float sq[128];
    if (threadIdx.x < 128) { s[threadIdx.x] = 0.f; sq[threadIdx.x] = 0.f; }
    __syncthreads();
    const int total = N_NODES * 128;
    for (int idx = blockIdx.x * blockDim.x + threadIdx.x; idx < total; idx += gstride()) {
        float v = x[idx];
        int f = idx & 127;
        atomicAdd(&s[f], v);
        atomicAdd(&sq[f], v * v);
    }
    __syncthreads();
    if (threadIdx.x < 128) {
        atomicAdd(&stats[threadIdx.x],       s[threadIdx.x]);
        atomicAdd(&stats[128 + threadIdx.x], sq[threadIdx.x]);
    }
}

__global__ void k_bn_relu128(const float* __restrict__ x, const float* __restrict__ stats,
                             const float* __restrict__ gamma, const float* __restrict__ beta,
                             float* __restrict__ y) {
    const float invn = 1.0f / (float)N_NODES;
    const int total = N_NODES * 128;
    for (int idx = blockIdx.x * blockDim.x + threadIdx.x; idx < total; idx += gstride()) {
        int f = idx & 127;
        float mean = stats[f] * invn;
        float var  = stats[128 + f] * invn - mean * mean;
        float v = (x[idx] - mean) * rsqrtf(var + BN_EPS) * gamma[f] + beta[f];
        y[idx] = v > 0.f ? v : 0.f;
    }
}

// ---------- pooling ----------
__global__ void k_count(const int* __restrict__ batch, float* __restrict__ counts) {
    for (int i = blockIdx.x * blockDim.x + threadIdx.x; i < N_NODES; i += gstride())
        atomicAdd(&counts[batch[i]], 1.0f);
}

__global__ void k_pool_scatter(const float* __restrict__ h, const int* __restrict__ batch,
                               float* __restrict__ out) {
    const int total = N_NODES * 256;
    for (int idx = blockIdx.x * blockDim.x + threadIdx.x; idx < total; idx += gstride()) {
        int i = idx >> 8, f = idx & 255;
        atomicAdd(&out[batch[i] * 256 + f], h[idx]);
    }
}

__global__ void k_pool_div(float* __restrict__ out, const float* __restrict__ counts) {
    const int total = N_GRAPHS * 256;
    for (int idx = blockIdx.x * blockDim.x + threadIdx.x; idx < total; idx += gstride())
        out[idx] = out[idx] / fmaxf(counts[idx >> 8], 1.0f);
}

// ---------- host-side launcher ----------
static inline int nblk(long long total, int cap) {
    long long b = (total + TPB - 1) / TPB;
    return (int)(b < cap ? b : cap);
}

extern "C" void kernel_launch(void* const* d_in, const int* in_sizes, int n_in,
                              void* d_out, int out_size, void* d_ws, size_t ws_size,
                              hipStream_t stream) {
    const float* x    = (const float*)d_in[0];
    const int*   edge = (const int*)d_in[1];
    const int*   src  = edge;
    const int*   tgt  = edge + N_EDGES;
    const int*   batch = (const int*)d_in[2];
    const float* W1 = (const float*)d_in[3];
    const float* b1 = (const float*)d_in[4];
    const float* W2 = (const float*)d_in[5];
    const float* b2 = (const float*)d_in[6];
    const float* W3 = (const float*)d_in[7];
    const float* b3 = (const float*)d_in[8];
    const float* g1 = (const float*)d_in[9];
    const float* be1 = (const float*)d_in[10];
    const float* g2 = (const float*)d_in[11];
    const float* be2 = (const float*)d_in[12];
    float* out = (float*)d_out;

    // workspace layout (bytes)
    char* ws = (char*)d_ws;
    float* hA    = (float*)(ws);                              // 100000*256*4 = 102,400,000
    float* hB    = (float*)(ws + 102400000);                  // 102,400,000
    f16*   Hh    = (f16*)  (ws + 204800000);                  // 100000*128*2 = 25,600,000
    f16*   Wt    = (f16*)  (ws + 230400000);                  // 256*128*2 = 65,536
    float* dinv  = (float*)(ws + 230465536);                  // 100000*4
    float* stats = (float*)(ws + 230865536);                  // 2*256*4
    float* counts= (float*)(ws + 230867584);                  // 4096*4

    const int CAP = 65536;
    const int GEMM_BLOCKS = (N_NODES / 16 + 7) / 8;           // 782 (8 row-strips per block)

    // degrees (self-loop => init 1.0), then dinv = rsqrt(deg)
    k_fill_f32<<<nblk(N_NODES, CAP), TPB, 0, stream>>>(dinv, 1.0f, N_NODES);
    k_deg<<<nblk(N_EDGES, CAP), TPB, 0, stream>>>(tgt, dinv);
    k_rsqrt_inplace<<<nblk(N_NODES, CAP), TPB, 0, stream>>>(dinv, N_NODES);

    // ---- layer 1: h = relu(bn(gcn(x, W1, b1))) ----
    k_cvt_f16<<<nblk((long long)N_NODES * 128, CAP), TPB, 0, stream>>>(x, Hh, N_NODES * 128);
    k_wt_cvt<<<nblk(K_IN * 128, CAP), TPB, 0, stream>>>(W1, Wt, 128);
    k_gemm_wmma_lds<128><<<GEMM_BLOCKS, TPB, 128 * K_IN * sizeof(f16), stream>>>(
        Hh, Wt, dinv, b1, hA /*hw*/, hB /*agg*/, N_NODES);
    k_edge_scatter<<<nblk((long long)N_EDGES * 128, CAP), TPB, 0, stream>>>(src, tgt, hA, dinv, hB, 128);
    k_fill_f32<<<1, TPB, 0, stream>>>(stats, 0.0f, 256);
    k_bn_stats128<<<2048, TPB, 0, stream>>>(hB, stats);
    k_bn_relu128<<<nblk((long long)N_NODES * 128, CAP), TPB, 0, stream>>>(hB, stats, g1, be1, hA);

    // ---- layer 2 ----
    k_cvt_f16<<<nblk((long long)N_NODES * 128, CAP), TPB, 0, stream>>>(hA, Hh, N_NODES * 128);
    k_wt_cvt<<<nblk(K_IN * 128, CAP), TPB, 0, stream>>>(W2, Wt, 128);
    k_gemm_wmma_lds<128><<<GEMM_BLOCKS, TPB, 128 * K_IN * sizeof(f16), stream>>>(
        Hh, Wt, dinv, b2, hB /*hw*/, hA /*agg*/, N_NODES);
    k_edge_scatter<<<nblk((long long)N_EDGES * 128, CAP), TPB, 0, stream>>>(src, tgt, hB, dinv, hA, 128);
    k_fill_f32<<<1, TPB, 0, stream>>>(stats, 0.0f, 256);
    k_bn_stats128<<<2048, TPB, 0, stream>>>(hA, stats);
    k_bn_relu128<<<nblk((long long)N_NODES * 128, CAP), TPB, 0, stream>>>(hA, stats, g2, be2, hB);

    // ---- layer 3 (no BN/ReLU), Fout = 256 ----
    k_cvt_f16<<<nblk((long long)N_NODES * 128, CAP), TPB, 0, stream>>>(hB, Hh, N_NODES * 128);
    k_wt_cvt<<<nblk(K_IN * 256, CAP), TPB, 0, stream>>>(W3, Wt, 256);
    k_gemm_wmma_lds<256><<<GEMM_BLOCKS, TPB, 256 * K_IN * sizeof(f16), stream>>>(
        Hh, Wt, dinv, b3, hA /*hw*/, hB /*agg*/, N_NODES);
    k_edge_scatter<<<nblk((long long)N_EDGES * 256, CAP), TPB, 0, stream>>>(src, tgt, hA, dinv, hB, 256);

    // ---- global mean pool -> d_out ----
    k_fill_f32<<<nblk(N_GRAPHS * 256, CAP), TPB, 0, stream>>>(out, 0.0f, N_GRAPHS * 256);
    k_fill_f32<<<nblk(N_GRAPHS, CAP), TPB, 0, stream>>>(counts, 0.0f, N_GRAPHS);
    k_count<<<nblk(N_NODES, CAP), TPB, 0, stream>>>(batch, counts);
    k_pool_scatter<<<nblk((long long)N_NODES * 256, CAP), TPB, 0, stream>>>(hB, batch, out);
    k_pool_div<<<nblk(N_GRAPHS * 256, CAP), TPB, 0, stream>>>(out, counts);
}
